// generator_78666620993539
// MI455X (gfx1250) — matrix operand order repaired and checked by reading, not compiled, not run on hardware
//
#include <hip/hip_runtime.h>
#include <hip/hip_bf16.h>
#include <math.h>

// ---------------- problem constants (match reference) ----------------
#define NLN   16384
#define NDN   4096
#define NMN   2048
#define EEDGE 262144
#define FEAT  256
#define HID   64
#define OUTF  128
#define ITEMS 4096
#define SROWS 2048

typedef __attribute__((ext_vector_type(16))) __bf16        v16bf;
typedef __attribute__((ext_vector_type(8)))  float         v8f;
typedef __attribute__((ext_vector_type(8)))  unsigned int  v8u;

__device__ __forceinline__ unsigned short f2bf(float f) {
  // round-to-nearest-even fp32 -> bf16
  unsigned int u = __float_as_uint(f);
  unsigned int r = u + 0x7fffu + ((u >> 16) & 1u);
  return (unsigned short)(r >> 16);
}

// ---------------------------------------------------------------------
// bf16 WMMA GEMM:  C[M,N] = act( (rowScale .* A)[M,K] @ B[K,N] + bias )
// A,B,C fp32 row-major. Requires M%128==0, N%64==0, K%32==0 (all sizes
// in this workload satisfy this). ACT: 0=none 1=relu 2=sigmoid(fast rcp).
// Block = 256 threads (8 waves, wave32); block tile 128x64; wave 32x32.
// Software pipeline: double-buffered LDS, global loads for tile i+1
// issued before the ds_load/WMMA of tile i; one barrier per K-step.
// ---------------------------------------------------------------------
template <int ACT, bool HAS_BIAS, bool HAS_SCALE>
__global__ __launch_bounds__(256)
void gemm_bf16_wmma(const float* __restrict__ A, int lda,
                    const float* __restrict__ B, int ldb,
                    const float* __restrict__ bias,
                    const float* __restrict__ rowScale,
                    float* __restrict__ C, int ldc, int K) {
  __shared__ unsigned short lA[2][128 * 34];   // [buf][row][k], +2 pad
  __shared__ unsigned short lB[2][64 * 34];    // [buf][n][k] (transposed), +2 pad

  const int t    = threadIdx.x;
  const int lane = t & 31;
  const int wid  = t >> 5;
  const int wm   = wid >> 1;          // wave row 0..3 (32 rows each)
  const int wn   = wid & 1;           // wave col 0..1 (32 cols each)
  const int rowBlk = blockIdx.y * 128;
  const int colBlk = blockIdx.x * 64;

  v8f acc[2][2] = {};

  // A staging map: thread -> (row = t>>1, k half = (t&1)*16)
  const int arow = t >> 1;
  const int aks  = (t & 1) * 16;
  // B staging map: thread -> (k = t>>3, n octet = (t&7)*8)
  const int bk  = t >> 3;
  const int bns = (t & 7) * 8;

  const float ascale = HAS_SCALE ? rowScale[rowBlk + arow] : 1.0f;
  const float* Ap = A + (long long)(rowBlk + arow) * lda + aks;
  const float* Bp = B + (long long)bk * ldb + colBlk + bns;

  const int frow = lane & 15;   // fragment row (A) / col (B)
  const int kh   = lane >> 4;   // lane-group K half

  float aReg[16];
  float bReg[8];

  auto loadTile = [&]() {
    #pragma unroll
    for (int j = 0; j < 16; j += 4) {
      float4 v = *(const float4*)(Ap + j);
      aReg[j] = v.x; aReg[j + 1] = v.y; aReg[j + 2] = v.z; aReg[j + 3] = v.w;
    }
    #pragma unroll
    for (int j = 0; j < 8; j += 4) {
      float4 v = *(const float4*)(Bp + j);
      bReg[j] = v.x; bReg[j + 1] = v.y; bReg[j + 2] = v.z; bReg[j + 3] = v.w;
    }
  };
  auto storeTile = [&](int buf) {
    // backend merges adjacent b16 stores into dword LDS stores
    unsigned short* d = &lA[buf][arow * 34 + aks];
    #pragma unroll
    for (int j = 0; j < 16; ++j)
      d[j] = f2bf(HAS_SCALE ? aReg[j] * ascale : aReg[j]);
    // B: transposed scatter, 8 b16 stores
    #pragma unroll
    for (int j = 0; j < 8; ++j)
      lB[buf][(bns + j) * 34 + bk] = f2bf(bReg[j]);
  };

  // ---- preload tile 0 ----
  loadTile();
  storeTile(0);
  __syncthreads();

  const int nIter = K >> 5;
  for (int it = 0; it < nIter; ++it) {
    const int cur = it & 1, nxt = cur ^ 1;
    const bool hasNext = (it + 1) < nIter;
    if (hasNext) {
      Ap += 32;
      Bp += (long long)32 * ldb;
      // issue next tile's global loads first so they overlap ds/WMMA below
      loadTile();
      __builtin_prefetch(Ap + 64, 0, 1);   // global_prefetch_b8, 2 tiles ahead
    }

    // ---- build fragments (documented 16-bit WMMA lane layouts) ----
    v16bf afrag[2], bfrag[2];
    #pragma unroll
    for (int h = 0; h < 2; ++h) {
      v8u au;
      const unsigned short* base = &lA[cur][(wm * 32 + h * 16 + frow) * 34];
      #pragma unroll
      for (int v = 0; v < 8; ++v) {
        const int ks = (v < 4 ? 2 * v : 16 + 2 * (v - 4)) + 8 * kh;
        au[v] = *(const unsigned int*)(base + ks);   // packed pair (K,K+1)
      }
      afrag[h] = __builtin_bit_cast(v16bf, au);
    }
    #pragma unroll
    for (int h = 0; h < 2; ++h) {
      v8u bu;
      const unsigned short* base = &lB[cur][(wn * 32 + h * 16 + frow) * 34];
      #pragma unroll
      for (int v = 0; v < 8; ++v)
        bu[v] = *(const unsigned int*)(base + (v < 4 ? 2 * v : 16 + 2 * (v - 4)) + 8 * kh);
      bfrag[h] = __builtin_bit_cast(v16bf, bu);
    }

    #pragma unroll
    for (int i = 0; i < 2; ++i)
      #pragma unroll
      for (int j = 0; j < 2; ++j)
        acc[i][j] = __builtin_amdgcn_wmma_f32_16x16x32_bf16(
            false, afrag[i], false, bfrag[j], (short)0, acc[i][j],
            false, false);

    if (hasNext) storeTile(nxt);
    __syncthreads();
  }

  // ---- epilogue: bias + activation + store (C/D layout) ----
  const int crow0 = rowBlk + wm * 32;
  const int ccol0 = colBlk + wn * 32;
  const int cl    = lane & 15;
  const int chalf = lane >> 4;
  #pragma unroll
  for (int i = 0; i < 2; ++i) {
    #pragma unroll
    for (int j = 0; j < 2; ++j) {
      const int col = ccol0 + j * 16 + cl;
      const float bv = HAS_BIAS ? bias[col] : 0.0f;
      #pragma unroll
      for (int r = 0; r < 8; ++r) {
        const int row = crow0 + i * 16 + chalf * 8 + r;
        float x = acc[i][j][r] + bv;
        if (ACT == 1) x = fmaxf(x, 0.0f);
        if (ACT == 2) x = __builtin_amdgcn_rcpf(1.0f + __expf(-x));
        C[(long long)row * ldc + col] = x;
      }
    }
  }
}

// ------------------------- support kernels ---------------------------
__global__ void fill_zero(float* __restrict__ p, int n) {
  int i = blockIdx.x * blockDim.x + threadIdx.x;
  if (i < n) p[i] = 0.0f;
}

__global__ void edge_degrees(const int* __restrict__ src, const int* __restrict__ dst,
                             float* __restrict__ degS, float* __restrict__ degD) {
  int e = blockIdx.x * blockDim.x + threadIdx.x;
  if (e >= EEDGE) return;
  atomicAdd(&degS[src[e]], 1.0f);
  atomicAdd(&degD[dst[e]], 1.0f);
}

__global__ void finalize_rsqrt(float* __restrict__ p, int n) {
  int i = blockIdx.x * blockDim.x + threadIdx.x;
  if (i < n) p[i] = rsqrtf(fmaxf(p[i], 1.0f));
}

// acc[i, f] = b[t1*w + f] + b[t2*w + f]   (bias of both incoming gconvs)
__global__ void init_bias2(float* __restrict__ acc, const float* __restrict__ b,
                           int t1, int t2, int total, int width) {
  int i = blockIdx.x * blockDim.x + threadIdx.x;
  if (i >= total) return;
  int f = i % width;
  acc[i] = b[t1 * width + f] + b[t2 * width + f];
}

// acc[dst[e], :] += rs_in[dst[e]] * Y[src[e], :]   (rs_out folded into GEMM)
__global__ void scatter_edges(const int* __restrict__ src, const int* __restrict__ dst,
                              const float* __restrict__ rsin,
                              const float* __restrict__ Y, float* __restrict__ acc,
                              int width) {
  const int qpr = width >> 2;                 // float4 per row
  long long gid = (long long)blockIdx.x * blockDim.x + threadIdx.x;
  if (gid >= (long long)EEDGE * qpr) return;
  const int e = (int)(gid / qpr);
  const int q = (int)(gid % qpr);
  const int s = src[e], d = dst[e];
  const float c = rsin[d];
  const float4 y = *(const float4*)(Y + (long long)s * width + q * 4);
  float* o = acc + (long long)d * width + q * 4;
  atomicAdd(o + 0, c * y.x); atomicAdd(o + 1, c * y.y);
  atomicAdd(o + 2, c * y.z); atomicAdd(o + 3, c * y.w);
}

// fake[i, :] = H[leftIndex+i, :] / max(sum|H[leftIndex+i,:]|, 1e-12)
__global__ __launch_bounds__(128)
void l1norm_rows(const float* __restrict__ H, const int* __restrict__ leftIndexP,
                 float* __restrict__ outFake) {
  __shared__ float red[4];
  const int i = blockIdx.x;           // 0..SROWS-1
  const int f = threadIdx.x;          // 0..OUTF-1
  const int row = leftIndexP[0] + i;
  const float v = H[(long long)row * OUTF + f];
  float a = fabsf(v);
  #pragma unroll
  for (int off = 16; off > 0; off >>= 1) a += __shfl_down(a, off, 32);
  if ((f & 31) == 0) red[f >> 5] = a;
  __syncthreads();
  const float s = fmaxf(red[0] + red[1] + red[2] + red[3], 1e-12f);
  outFake[(long long)i * OUTF + f] = v / s;
}

__global__ void build_xcat(const float* __restrict__ Adj, const float* __restrict__ fake,
                           float* __restrict__ xcat) {
  const int rowW = ITEMS + OUTF;
  long long gid = (long long)blockIdx.x * blockDim.x + threadIdx.x;
  if (gid >= (long long)SROWS * rowW) return;
  const int i = (int)(gid / rowW), c = (int)(gid % rowW);
  xcat[gid] = (c < ITEMS) ? Adj[(long long)i * ITEMS + c]
                          : fake[(long long)i * OUTF + (c - ITEMS)];
}

// ------------------------------ launch -------------------------------
extern "C" void kernel_launch(void* const* d_in, const int* in_sizes, int n_in,
                              void* d_out, int out_size, void* d_ws, size_t ws_size,
                              hipStream_t stream) {
  const float* h_L = (const float*)d_in[0];
  const float* h_D = (const float*)d_in[1];
  const float* h_M = (const float*)d_in[2];
  const float* Adj = (const float*)d_in[3];
  const int* srcP[6] = {(const int*)d_in[4],  (const int*)d_in[6],  (const int*)d_in[8],
                        (const int*)d_in[10], (const int*)d_in[12], (const int*)d_in[14]};
  const int* dstP[6] = {(const int*)d_in[5],  (const int*)d_in[7],  (const int*)d_in[9],
                        (const int*)d_in[11], (const int*)d_in[13], (const int*)d_in[15]};
  const float* W1 = (const float*)d_in[16]; const float* b1 = (const float*)d_in[17];
  const float* W2 = (const float*)d_in[18]; const float* b2 = (const float*)d_in[19];
  const float* W3 = (const float*)d_in[20]; const float* b3 = (const float*)d_in[21];
  const float* f1w = (const float*)d_in[22]; const float* f1b = (const float*)d_in[23];
  const float* f2w = (const float*)d_in[24]; const float* f2b = (const float*)d_in[25];
  const float* f3w = (const float*)d_in[26]; const float* f3b = (const float*)d_in[27];
  const float* f4w = (const float*)d_in[28]; const float* f4b = (const float*)d_in[29];
  const int* leftIndexP = (const int*)d_in[31];
  float* out = (float*)d_out;                // [SROWS*OUTF fake | SROWS*ITEMS x]

  // ---- workspace carve-up (floats) ----
  float* ws = (float*)d_ws;
  size_t off = 0;
  auto carve = [&](size_t n) { float* p = ws + off; off += n; return p; };
  float* bufL0 = carve((size_t)NLN * FEAT);
  float* bufL1 = carve((size_t)NLN * FEAT);
  float* bufD0 = carve((size_t)NDN * FEAT);
  float* bufD1 = carve((size_t)NDN * FEAT);
  float* bufM0 = carve((size_t)NMN * FEAT);
  float* bufM1 = carve((size_t)NMN * FEAT);
  float* Ybuf  = carve((size_t)NLN * OUTF);
  const int nsrc[6] = {NLN, NDN, NLN, NMN, NDN, NMN};
  const int ndst[6] = {NDN, NLN, NMN, NLN, NMN, NDN};
  float* rsOut[6]; float* rsIn[6];
  float* rsBase = ws + off;
  for (int t = 0; t < 6; ++t) { rsOut[t] = carve(nsrc[t]); rsIn[t] = carve(ndst[t]); }
  const int rsTotal = (int)((ws + off) - rsBase);
  float* xcat = carve((size_t)SROWS * (ITEMS + OUTF));
  float* m1   = carve((size_t)SROWS * 256);
  float* m2   = carve((size_t)SROWS * 512);
  float* m3   = carve((size_t)SROWS * 1024);
  (void)ws_size; (void)in_sizes; (void)n_in; (void)out_size;

  // act: 0=none 1=relu 2=sigmoid; rowScale path has no bias (projections)
  auto gemm = [&](const float* A, int lda, const float* B, int ldb,
                  const float* bias, const float* rowScale,
                  float* C, int ldc, int M, int N, int K, int act) {
    dim3 g(N / 64, M / 128);
    if (rowScale)
      gemm_bf16_wmma<0, false, true ><<<g, 256, 0, stream>>>(A, lda, B, ldb, nullptr, rowScale, C, ldc, K);
    else if (act == 1)
      gemm_bf16_wmma<1, true,  false><<<g, 256, 0, stream>>>(A, lda, B, ldb, bias, nullptr, C, ldc, K);
    else if (act == 2)
      gemm_bf16_wmma<2, true,  false><<<g, 256, 0, stream>>>(A, lda, B, ldb, bias, nullptr, C, ldc, K);
    else
      gemm_bf16_wmma<0, true,  false><<<g, 256, 0, stream>>>(A, lda, B, ldb, bias, nullptr, C, ldc, K);
  };

  // ---- degree normalizers (shared by all 3 layers) ----
  fill_zero<<<(rsTotal + 255) / 256, 256, 0, stream>>>(rsBase, rsTotal);
  for (int t = 0; t < 6; ++t)
    edge_degrees<<<EEDGE / 256, 256, 0, stream>>>(srcP[t], dstP[t], rsOut[t], rsIn[t]);
  finalize_rsqrt<<<(rsTotal + 255) / 256, 256, 0, stream>>>(rsBase, rsTotal);

  // ---- one hetero GraphConv layer (project-first) ----
  auto run_layer = [&](const float* xL, const float* xD, const float* xM, int inW,
                       const float* W, const float* b, int outW,
                       float* oL, float* oD, float* oM) {
    const float* srcX[6]  = {xL, xD, xL, xM, xD, xM};
    float*       dstA[6]  = {oD, oL, oM, oL, oM, oD};
    // bias init: oD <- b0+b5, oL <- b1+b3, oM <- b2+b4
    init_bias2<<<((NDN * outW) + 255) / 256, 256, 0, stream>>>(oD, b, 0, 5, NDN * outW, outW);
    init_bias2<<<((NLN * outW) + 255) / 256, 256, 0, stream>>>(oL, b, 1, 3, NLN * outW, outW);
    init_bias2<<<((NMN * outW) + 255) / 256, 256, 0, stream>>>(oM, b, 2, 4, NMN * outW, outW);
    for (int t = 0; t < 6; ++t) {
      // Y = (rs_out .* X_src) @ W[t]
      gemm(srcX[t], inW, W + (size_t)t * inW * outW, outW,
           nullptr, rsOut[t], Ybuf, outW, nsrc[t], outW, inW, 0);
      // acc[dst] += rs_in[dst] * Y[src]
      long long thr = (long long)EEDGE * (outW / 4);
      scatter_edges<<<(unsigned)((thr + 255) / 256), 256, 0, stream>>>(
          srcP[t], dstP[t], rsIn[t], Ybuf, dstA[t], outW);
    }
  };

  run_layer(h_L,   h_D,   h_M,   FEAT, W1, b1, HID,  bufL0, bufD0, bufM0);
  run_layer(bufL0, bufD0, bufM0, HID,  W2, b2, HID,  bufL1, bufD1, bufM1);
  run_layer(bufL1, bufD1, bufM1, HID,  W3, b3, OUTF, bufL0, bufD0, bufM0);

  // ---- L1-normalized embedding slice -> output[0 .. SROWS*OUTF) ----
  l1norm_rows<<<SROWS, 128, 0, stream>>>(bufL0, leftIndexP, out);

  // ---- decoder MLP ----
  {
    long long thr = (long long)SROWS * (ITEMS + OUTF);
    build_xcat<<<(unsigned)((thr + 255) / 256), 256, 0, stream>>>(Adj, out, xcat);
  }
  gemm(xcat, ITEMS + OUTF, f1w, 256,  f1b, nullptr, m1, 256,  SROWS, 256,  ITEMS + OUTF, 1);
  gemm(m1,   256,          f2w, 512,  f2b, nullptr, m2, 512,  SROWS, 512,  256,          1);
  gemm(m2,   512,          f3w, 1024, f3b, nullptr, m3, 1024, SROWS, 1024, 512,          1);
  gemm(m3,   1024,         f4w, ITEMS, f4b, nullptr, out + (size_t)SROWS * OUTF, ITEMS,
       SROWS, ITEMS, 1024, 2);
}
